// AutoregressiveLSTM_29764123361431
// MI455X (gfx1250) — matrix operand order repaired and checked by reading, not compiled, not run on hardware
//
#include <hip/hip_runtime.h>
#include <hip/hip_bf16.h>
#include <math.h>

// Problem constants
#define B_   256
#define T_   128
#define COV_ 32
#define E_   64
#define H_   1024
#define K0_  1120   // COV+E + H   (layer0 concat input)
#define K1_  2048   // H + H       (layer1 concat input)
#define NG_  4096   // 4*H gate columns

typedef __attribute__((ext_vector_type(16))) __bf16 bf16x16;
typedef __attribute__((ext_vector_type(8)))  float  f32x8;
typedef int v4i_ __attribute__((vector_size(16)));

union Frag { bf16x16 v; uint4 q[2]; };

#if defined(__gfx1250__) && __has_builtin(__builtin_amdgcn_global_load_async_to_lds_b128) && __has_builtin(__builtin_amdgcn_s_wait_asynccnt)
#define HAVE_ASYNC_LDS 1
#define GLOBAL_V4(p) ((__attribute__((address_space(1))) v4i_*)(p))
#define LDS_V4(p)    ((__attribute__((address_space(3))) v4i_*)(p))
#else
#define HAVE_ASYNC_LDS 0
#endif

static __device__ __forceinline__ unsigned short f2bf(float f) {
    union { float f; unsigned u; } v; v.f = f;
    unsigned r = v.u + 0x7FFFu + ((v.u >> 16) & 1u);   // round-to-nearest-even
    return (unsigned short)(r >> 16);
}
static __device__ __forceinline__ float sigm(float x) {
    return 1.0f / (1.0f + expf(-x));
}

// ---------------- weight packing: Wp[k][n] = (k<Kih ? Wih[n][k] : Whh[n][k-Kih]) as bf16
__global__ void pack_w_kernel(const float* __restrict__ Wih, const float* __restrict__ Whh,
                              int Kih, unsigned short* __restrict__ Wp, int K) {
    size_t idx = (size_t)blockIdx.x * 256 + threadIdx.x;
    if (idx >= (size_t)K * NG_) return;
    int k = (int)(idx >> 12);           // /4096
    int n = (int)(idx & (NG_ - 1));
    float w = (k < Kih) ? Wih[(size_t)n * Kih + k]
                        : Whh[(size_t)n * H_ + (k - Kih)];
    Wp[idx] = f2bf(w);
}

__global__ void bias_add_kernel(const float* __restrict__ a, const float* __restrict__ b,
                                float* __restrict__ out) {
    int i = blockIdx.x * 256 + threadIdx.x;
    if (i < NG_) out[i] = a[i] + b[i];
}

__global__ void zero_f32_kernel(float* __restrict__ p, int n) {
    int i = blockIdx.x * 256 + threadIdx.x;
    if (i < n) p[i] = 0.0f;
}

// zero a [B_ x cols] bf16 sub-region with row stride ld starting at column off
__global__ void zero_bf16_region_kernel(unsigned short* __restrict__ dst, int ld, int off, int cols) {
    int i = blockIdx.x * 256 + threadIdx.x;
    if (i >= B_ * cols) return;
    int b = i / cols, c = i % cols;
    dst[(size_t)b * ld + off + c] = 0;
}

// ---------------- per-step input prep: x = [cov | select(mask, mu*embW+embb, label)] -> bf16
__global__ void prep_kernel(const float* __restrict__ inputs, const float* __restrict__ labels,
                            const unsigned char* __restrict__ mask,
                            const float* __restrict__ embW, const float* __restrict__ embb,
                            const float* __restrict__ mubuf,
                            unsigned short* __restrict__ X, int t) {
    int b = blockIdx.x;
    int tid = threadIdx.x;                      // 96 threads
    float v;
    if (tid < COV_) {
        v = inputs[((size_t)b * T_ + t) * COV_ + tid];
    } else {
        int e = tid - COV_;
        float lbl = labels[((size_t)b * T_ + t) * E_ + e];
        float mue = mubuf[b] * embW[e] + embb[e];
        bool m = mask[(size_t)b * T_ + t] != 0;
        v = m ? mue : lbl;
    }
    X[(size_t)b * K0_ + tid] = f2bf(v);
}

// ---------------- fused GEMM (bf16 WMMA, f32 accum) + LSTM cell
// Block: 128 threads (4 waves). Block tile: M=32 rows x 64 h-cols (x 4 gates).
// Wave w = n-subtile (16 h cols); each wave owns BOTH 16-row M subtiles and all 4 gates:
// 8 accumulators of f32x8. LDS double-buffered; staging uses async global->LDS when available.
// A: [B_][K] bf16, W: [K][4096] bf16 (K-major), bias: [4096] f32, C: [B_][H_] f32 in/out
__global__ __launch_bounds__(128)
void lstm_gemm_cell_kernel(const unsigned short* __restrict__ A, int K,
                           const unsigned short* __restrict__ W,
                           const float* __restrict__ bias,
                           float* __restrict__ C,
                           unsigned short* __restrict__ hdst0, int ld0, int off0,
                           unsigned short* __restrict__ hdst1, int ld1, int off1,
                           float* __restrict__ hf32) {
    __shared__ __align__(16) unsigned short As[2][32][40];    // 32 rows x 32 K (+pad)
    __shared__ __align__(16) unsigned short Bs[2][32][264];   // 32 K x 256 N (4 gates x 64) (+pad)

    const int tid   = threadIdx.x;
    const int wave  = tid >> 5;          // 0..3 = N subtile
    const int lane  = tid & 31;
    const int m0    = blockIdx.y * 32;   // 8 M blocks
    const int n0    = blockIdx.x * 64;   // 16 N blocks (h columns)
    const int KT    = K >> 5;

    // cooperative-load indices (128 threads)
    const int a_row = tid >> 2, a_seg = tid & 3;            // 128 x 16B for A chunk

    // issue one K-chunk's staging into LDS buffer `buf`
    auto stage = [&](int kt, int buf) {
        const int k0 = kt * 32;
        const unsigned short* gA = A + (size_t)(m0 + a_row) * K + k0 + a_seg * 8;
#if HAVE_ASYNC_LDS
        __builtin_amdgcn_global_load_async_to_lds_b128(
            GLOBAL_V4(gA), LDS_V4(&As[buf][a_row][a_seg * 8]), 0, 0);
#else
        *(uint4*)&As[buf][a_row][a_seg * 8] = *(const uint4*)gA;
#endif
        #pragma unroll
        for (int r = 0; r < 8; ++r) {                       // 1024 x 16B for B chunk
            int idx   = tid + r * 128;
            int kk    = idx >> 5;
            int piece = idx & 31;
            int gate  = piece >> 3;
            int offp  = piece & 7;
            const unsigned short* gB = W + (size_t)(k0 + kk) * NG_ + gate * H_ + n0 + offp * 8;
#if HAVE_ASYNC_LDS
            __builtin_amdgcn_global_load_async_to_lds_b128(
                GLOBAL_V4(gB), LDS_V4(&Bs[buf][kk][gate * 64 + offp * 8]), 0, 0);
#else
            *(uint4*)&Bs[buf][kk][gate * 64 + offp * 8] = *(const uint4*)gB;
#endif
        }
    };

    f32x8 acc[2][4] = {};

    stage(0, 0);
    int cur = 0;
    for (int kt = 0; kt < KT; ++kt) {
#if HAVE_ASYNC_LDS
        __builtin_amdgcn_s_wait_asynccnt(0);
#endif
        __syncthreads();                       // buf[cur] ready; buf[cur^1] readers done
        if (kt + 1 < KT) stage(kt + 1, cur ^ 1);

        // A fragments (16x32 bf16): lanes 0-15 -> K 0-7,16-23 ; lanes 16-31 -> K 8-15,24-31
        Frag a[2];
        {
            int arow = (lane & 15);
            int ak   = (lane >> 4) * 8;
            a[0].q[0] = *(const uint4*)&As[cur][arow][ak];
            a[0].q[1] = *(const uint4*)&As[cur][arow][ak + 16];
            a[1].q[0] = *(const uint4*)&As[cur][arow + 16][ak];
            a[1].q[1] = *(const uint4*)&As[cur][arow + 16][ak + 16];
        }
        // B fragments (32x16 bf16): lane&15 = N; lanes 0-15 -> K 0-7,16-23 ; lanes 16-31 -> K 8-15,24-31
        Frag bfr[4];
        #pragma unroll
        for (int g = 0; g < 4; ++g) {
            const int bn = g * 64 + wave * 16 + (lane & 15);
            const int bk = (lane >> 4) * 8;
            #pragma unroll
            for (int s = 0; s < 8; ++s) {
                bfr[g].v[s]     = __builtin_bit_cast(__bf16, Bs[cur][bk + s][bn]);
                bfr[g].v[s + 8] = __builtin_bit_cast(__bf16, Bs[cur][16 + bk + s][bn]);
            }
        }
        #pragma unroll
        for (int mt = 0; mt < 2; ++mt)
            #pragma unroll
            for (int g = 0; g < 4; ++g)
                acc[mt][g] = __builtin_amdgcn_wmma_f32_16x16x32_bf16(
                    false, a[mt].v, false, bfr[g].v, (short)0, acc[mt][g], false, false);
        cur ^= 1;
    }

    // LSTM cell epilogue: C/D layout -> lane holds (M = r + 8*(lane>=16), N = lane&15), r=0..7
    const int nloc = lane & 15;
    const int mhi  = (lane >> 4) * 8;
    const int n    = n0 + wave * 16 + nloc;
    const float bi = bias[n];
    const float bf = bias[H_ + n];
    const float bg = bias[2 * H_ + n];
    const float bo = bias[3 * H_ + n];
    #pragma unroll
    for (int mt = 0; mt < 2; ++mt) {
        #pragma unroll
        for (int r = 0; r < 8; ++r) {
            int m = m0 + mt * 16 + mhi + r;
            size_t ci = (size_t)m * H_ + n;
            float gi = sigm(acc[mt][0][r] + bi);
            float gf = sigm(acc[mt][1][r] + bf);
            float gg = tanhf(acc[mt][2][r] + bg);
            float go = sigm(acc[mt][3][r] + bo);
            float c  = gf * C[ci] + gi * gg;
            C[ci]    = c;
            float h  = go * tanhf(c);
            hf32[ci] = h;
            unsigned short hb = f2bf(h);
            hdst0[(size_t)m * ld0 + off0 + n] = hb;
            if (hdst1) hdst1[(size_t)m * ld1 + off1 + n] = hb;
        }
    }
}

// ---------------- mu / sigma projection (hp is layer-interleaved: hp[2k]=h0[k], hp[2k+1]=h1[k])
__global__ __launch_bounds__(256)
void proj_kernel(const float* __restrict__ h0, const float* __restrict__ h1,
                 const float* __restrict__ muW, const float* __restrict__ mub,
                 const float* __restrict__ sigW, const float* __restrict__ sigb,
                 float* __restrict__ mus, float* __restrict__ sigs,
                 float* __restrict__ mubuf, int t) {
    int wave = threadIdx.x >> 5, lane = threadIdx.x & 31;
    int b = blockIdx.x * 8 + wave;
    float sm = 0.f, ss = 0.f;
    for (int k = lane; k < H_; k += 32) {
        float a0 = h0[(size_t)b * H_ + k];
        float a1 = h1[(size_t)b * H_ + k];
        sm += a0 * muW[2 * k]  + a1 * muW[2 * k + 1];
        ss += a0 * sigW[2 * k] + a1 * sigW[2 * k + 1];
    }
    #pragma unroll
    for (int off = 16; off > 0; off >>= 1) {
        sm += __shfl_down(sm, off, 32);
        ss += __shfl_down(ss, off, 32);
    }
    if (lane == 0) {
        float mu = sm + mub[0];
        float sg = ss + sigb[0];
        float sp = (sg > 20.f) ? sg : log1pf(expf(sg));
        mus[(size_t)b * T_ + t]  = mu;
        sigs[(size_t)b * T_ + t] = sp;
        mubuf[b] = mu;
    }
}

// ---------------- final (h, c) assembly into output
__global__ void finalize_kernel(const float* __restrict__ h0, const float* __restrict__ h1,
                                const float* __restrict__ c0, const float* __restrict__ c1,
                                float* __restrict__ out_hc) {
    size_t idx = (size_t)blockIdx.x * 256 + threadIdx.x;
    const size_t BH = (size_t)B_ * H_;
    if (idx < BH)            out_hc[idx] = h0[idx];
    else if (idx < 2 * BH)   out_hc[idx] = h1[idx - BH];
    else if (idx < 3 * BH)   out_hc[idx] = c0[idx - 2 * BH];
    else if (idx < 4 * BH)   out_hc[idx] = c1[idx - 3 * BH];
}

extern "C" void kernel_launch(void* const* d_in, const int* in_sizes, int n_in,
                              void* d_out, int out_size, void* d_ws, size_t ws_size,
                              hipStream_t stream) {
    (void)in_sizes; (void)n_in; (void)out_size; (void)ws_size;
    const float* inputs = (const float*)d_in[0];
    const float* labels = (const float*)d_in[1];
    const unsigned char* mask = (const unsigned char*)d_in[2];   // jax bool -> 1 byte
    const float* W_ih0 = (const float*)d_in[3];
    const float* W_hh0 = (const float*)d_in[4];
    const float* b_ih0 = (const float*)d_in[5];
    const float* b_hh0 = (const float*)d_in[6];
    const float* W_ih1 = (const float*)d_in[7];
    const float* W_hh1 = (const float*)d_in[8];
    const float* b_ih1 = (const float*)d_in[9];
    const float* b_hh1 = (const float*)d_in[10];
    const float* emb_W = (const float*)d_in[11];
    const float* emb_b = (const float*)d_in[12];
    const float* mu_W  = (const float*)d_in[13];
    const float* mu_b  = (const float*)d_in[14];
    const float* sig_W = (const float*)d_in[15];
    const float* sig_b = (const float*)d_in[16];

    // workspace carve-up (256B aligned)
    char* ws = (char*)d_ws;
    size_t off = 0;
    auto carve = [&](size_t bytes) -> char* {
        char* p = ws + off;
        off = (off + bytes + 255) & ~(size_t)255;
        return p;
    };
    unsigned short* Wp0   = (unsigned short*)carve((size_t)K0_ * NG_ * 2);
    unsigned short* Wp1   = (unsigned short*)carve((size_t)K1_ * NG_ * 2);
    float*          bias0 = (float*)carve(NG_ * 4);
    float*          bias1 = (float*)carve(NG_ * 4);
    unsigned short* X[2]  = { (unsigned short*)carve((size_t)B_ * K0_ * 2),
                              (unsigned short*)carve((size_t)B_ * K0_ * 2) };
    unsigned short* HC[2] = { (unsigned short*)carve((size_t)B_ * K1_ * 2),
                              (unsigned short*)carve((size_t)B_ * K1_ * 2) };
    float* c0    = (float*)carve((size_t)B_ * H_ * 4);
    float* c1    = (float*)carve((size_t)B_ * H_ * 4);
    float* h0f   = (float*)carve((size_t)B_ * H_ * 4);
    float* h1f   = (float*)carve((size_t)B_ * H_ * 4);
    float* mubuf = (float*)carve(B_ * 4);

    float* mus  = (float*)d_out;
    float* sigs = mus + (size_t)B_ * T_;
    float* out_hc = sigs + (size_t)B_ * T_;

    // one-time setup (re-done every call; deterministic)
    {
        size_t n0 = (size_t)K0_ * NG_;
        size_t n1 = (size_t)K1_ * NG_;
        pack_w_kernel<<<(unsigned)((n0 + 255) / 256), 256, 0, stream>>>(W_ih0, W_hh0, COV_ + E_, Wp0, K0_);
        pack_w_kernel<<<(unsigned)((n1 + 255) / 256), 256, 0, stream>>>(W_ih1, W_hh1, H_, Wp1, K1_);
        bias_add_kernel<<<16, 256, 0, stream>>>(b_ih0, b_hh0, bias0);
        bias_add_kernel<<<16, 256, 0, stream>>>(b_ih1, b_hh1, bias1);
        int nbh = (B_ * H_ + 255) / 256;
        zero_f32_kernel<<<nbh, 256, 0, stream>>>(c0, B_ * H_);
        zero_f32_kernel<<<nbh, 256, 0, stream>>>(c1, B_ * H_);
        zero_f32_kernel<<<1, 256, 0, stream>>>(mubuf, B_);
        // h0 region of X[0] (cols 96..1119) and h1 region of HC[0] (cols 1024..2047) start at zero
        zero_bf16_region_kernel<<<(B_ * H_ + 255) / 256, 256, 0, stream>>>(X[0], K0_, COV_ + E_, H_);
        zero_bf16_region_kernel<<<(B_ * H_ + 255) / 256, 256, 0, stream>>>(HC[0], K1_, H_, H_);
    }

    dim3 ggrid(16, 8);   // 16 N-blocks (64 h cols) x 8 M-blocks (32 rows)
    for (int t = 0; t < T_; ++t) {
        const int p = t & 1, q = p ^ 1;
        prep_kernel<<<B_, COV_ + E_, 0, stream>>>(inputs, labels, mask, emb_W, emb_b, mubuf, X[p], t);
        // layer 0: reads X[p], writes h0 -> X[q] (cols 96..) and HC[p] (cols 0..1023), c0 in-place
        lstm_gemm_cell_kernel<<<ggrid, 128, 0, stream>>>(
            X[p], K0_, Wp0, bias0, c0,
            X[q], K0_, COV_ + E_,
            HC[p], K1_, 0,
            h0f);
        // layer 1: reads HC[p], writes h1 -> HC[q] (cols 1024..2047), c1 in-place
        lstm_gemm_cell_kernel<<<ggrid, 128, 0, stream>>>(
            HC[p], K1_, Wp1, bias1, c1,
            HC[q], K1_, H_,
            (unsigned short*)nullptr, 0, 0,
            h1f);
        proj_kernel<<<B_ / 8, 256, 0, stream>>>(h0f, h1f, mu_W, mu_b, sig_W, sig_b,
                                                mus, sigs, mubuf, t);
    }
    finalize_kernel<<<(4 * B_ * H_) / 256, 256, 0, stream>>>(h0f, h1f, c0, c1, out_hc);
}
